// BaseCrystalModel_40192303956467
// MI455X (gfx1250) — matrix operand order repaired and checked by reading, not compiled, not run on hardware
//
#include <hip/hip_runtime.h>
#include <hip/hip_bf16.h>
#include <math.h>

// ---------------------------------------------------------------------------
// SchNet-style GNN forward for gfx1250 (MI455X).
// WMMA f32_16x16x32_f16 for all 128-wide GEMMs; fused edge kernel keeps the
// [E,128] filter tile entirely in registers/LDS (saves ~1.2 GB HBM traffic).
// ---------------------------------------------------------------------------

typedef __attribute__((ext_vector_type(16))) _Float16 v16h;
typedef __attribute__((ext_vector_type(8)))  _Float16 v8h;
typedef __attribute__((ext_vector_type(8)))  float    v8f;

#define N_NODES 50000
#define E_EDGES 800000
#define G_GRAPHS 1024
#define C_CH 128
#define FIN_F 16
#define FE_F 3
#define L_LAYERS 3
#define LOG2_CONST 0.6931471805599453f
#define BN_EPS 1e-5f
#define PI_OVER_CUTOFF 0.3141592653589793f

#define LDA 136   // padded f16 stride (272 B, 16B-aligned rows, dodges bank conflicts)

__device__ __forceinline__ float ssp(float x) {
  // shifted softplus: log(1+exp(x)) - log(2), overflow-safe
  float sp = (x > 20.0f) ? x : log1pf(__expf(x));
  return sp - LOG2_CONST;
}

__device__ __forceinline__ v16h make16(v8h lo, v8h hi) {
  v16h r;
#pragma unroll
  for (int i = 0; i < 8; ++i) { r[i] = lo[i]; r[i + 8] = hi[i]; }
  return r;
}

// A fragment, 16x32 f16 (ISA 7.12.2): lane L holds row L&15; K-halves split by L>>4.
__device__ __forceinline__ v16h load_a_frag(const _Float16* tile, int lane, int k0) {
  int row = lane & 15;
  int half = lane >> 4;
  const _Float16* p = tile + row * LDA + k0 + half * 8;
  v8h lo = *(const v8h*)(p);       // K = k0+half*8 .. +7
  v8h hi = *(const v8h*)(p + 16);  // K = k0+16+half*8 .. +7
  return make16(lo, hi);
}

// B fragment, 32x16 f16 from LDS weight stored transposed [N][K]: lane L holds
// column L&15; K = k0 + (L>>4)*8 + e (e<8), k0 + 16 + (L>>4)*8 + (e-8) (e>=8).
__device__ __forceinline__ v16h load_b_frag(const _Float16* wt, int lane, int n0, int k0) {
  int col = lane & 15;
  int kh  = lane >> 4;
  const _Float16* p = wt + (n0 + col) * LDA + k0 + kh * 8;
  v8h lo = *(const v8h*)(p);
  v8h hi = *(const v8h*)(p + 16);
  return make16(lo, hi);
}

// ---------------------------------------------------------------------------
// Generic node GEMM: out[r,:] = A[r,:K] @ W[K,128] + bias   (mode 0)
//                    out[r,:] = res[r,:] + ssp(acc + bias)  (mode 1)
// Block = 256 thr (8 waves), 64 rows x 128 cols per block, K in {16,128}.
// ---------------------------------------------------------------------------
__global__ __launch_bounds__(256) void gemm_bias_ssp_kernel(
    const float* __restrict__ A, const float* __restrict__ W,
    const float* __restrict__ bias, float* __restrict__ out,
    const float* __restrict__ res, int nrows, int K, int mode) {
  __shared__ __align__(16) _Float16 sA[64 * LDA];
  __shared__ __align__(16) _Float16 sW[128 * LDA];

  int tid = threadIdx.x;
  int wave = tid >> 5, lane = tid & 31;
  int row0 = blockIdx.x * 64;
  int Kp = (K + 31) & ~31;

  // Stage W transposed into LDS as f16: sW[n][k], zero-pad K to Kp.
  for (int idx = tid; idx < 128 * Kp; idx += 256) {
    int k = idx >> 7, n = idx & 127;
    sW[n * LDA + k] = (k < K) ? (_Float16)W[k * C_CH + n] : (_Float16)0.0f;
  }
  // Stage A tile as f16, zero-pad rows past nrows and K to Kp.
  for (int idx = tid; idx < 64 * Kp; idx += 256) {
    int m = idx / Kp, k = idx % Kp;
    int row = row0 + m;
    float v = (row < nrows && k < K) ? A[(long)row * K + k] : 0.0f;
    sA[m * LDA + k] = (_Float16)v;
  }
  __syncthreads();

  int mt = wave >> 1;             // row tile 0..3 (16 rows each)
  int nbase = (wave & 1) * 64;    // col half 0 / 64
  v8f acc[4] = {};

  for (int kk = 0; kk < Kp; kk += 32) {
    v16h a = load_a_frag(sA + mt * 16 * LDA, lane, kk);
#pragma unroll
    for (int t = 0; t < 4; ++t) {
      v16h b = load_b_frag(sW, lane, nbase + t * 16, kk);
      acc[t] = __builtin_amdgcn_wmma_f32_16x16x32_f16(
          false, a, false, b, (short)0, acc[t], false, false);
    }
  }

  // Epilogue: C layout -> M = r + (lane>>4)*8, N = lane&15
  int m_in = (lane >> 4) * 8;
  int nc = lane & 15;
#pragma unroll
  for (int t = 0; t < 4; ++t) {
    int ncol = nbase + t * 16 + nc;
    float bv = bias[ncol];
#pragma unroll
    for (int r = 0; r < 8; ++r) {
      int row = row0 + mt * 16 + m_in + r;
      if (row < nrows) {
        float v = acc[t][r] + bv;
        long o = (long)row * C_CH + ncol;
        out[o] = (mode == 1) ? (res[o] + ssp(v)) : v;
      }
    }
  }
}

// ---------------------------------------------------------------------------
// Fused edge kernel: per 64-edge tile
//   t    = ssp(edge_attr @ We1 + be1)          (VALU, K=3)
//   filt = t @ We2 + be2                       (WMMA from LDS)
//   agg[dst] += hin[src] * filt * env          (gather + f32 atomics)
// ---------------------------------------------------------------------------
__global__ __launch_bounds__(256) void edge_cfconv_kernel(
    const float* __restrict__ hin, const float* __restrict__ edge_attr,
    const float* __restrict__ edge_weight, const int* __restrict__ eidx,
    const float* __restrict__ We1, const float* __restrict__ be1,
    const float* __restrict__ We2, const float* __restrict__ be2,
    float* __restrict__ agg) {
  __shared__ __align__(16) _Float16 sT[64 * LDA];
  __shared__ __align__(16) _Float16 sW[128 * LDA];
  __shared__ float sWe1[FE_F * 128];
  __shared__ float sbe1[128];
  __shared__ int   sSrc[64];
  __shared__ int   sDst[64];
  __shared__ float sEnv[64];

  int tid = threadIdx.x, wave = tid >> 5, lane = tid & 31;
  long e0 = (long)blockIdx.x * 64;

  for (int i = tid; i < FE_F * 128; i += 256) sWe1[i] = We1[i];
  if (tid < 128) sbe1[tid] = be1[tid];
  if (tid < 64) {
    long e = e0 + tid;
    int s = 0, d = 0;
    float env = 0.0f;
    if (e < E_EDGES) {
      s = eidx[e];
      d = eidx[E_EDGES + e];
      env = 0.5f * (__cosf(edge_weight[e] * PI_OVER_CUTOFF) + 1.0f);
      __builtin_prefetch(&hin[(long)s * C_CH], 0, 1);  // global_prefetch_b8
    }
    sSrc[tid] = s; sDst[tid] = d; sEnv[tid] = env;
  }
  // Stage We2 transposed -> sW[n][k]
  for (int idx = tid; idx < 128 * 128; idx += 256) {
    int k = idx >> 7, n = idx & 127;
    sW[n * LDA + k] = (_Float16)We2[idx];
  }
  __syncthreads();

  // Filter MLP stage 1 (K=3) straight into the f16 A tile.
  for (int idx = tid; idx < 64 * 128; idx += 256) {
    int m = idx >> 7, c = idx & 127;
    long e = e0 + m;
    float v = 0.0f;
    if (e < E_EDGES) {
      float a0 = edge_attr[e * 3 + 0];
      float a1 = edge_attr[e * 3 + 1];
      float a2 = edge_attr[e * 3 + 2];
      v = ssp(a0 * sWe1[c] + a1 * sWe1[128 + c] + a2 * sWe1[256 + c] + sbe1[c]);
    }
    sT[m * LDA + c] = (_Float16)v;
  }
  __syncthreads();

  int mt = wave >> 1;
  int nbase = (wave & 1) * 64;
  v8f acc[4] = {};
  for (int kk = 0; kk < 128; kk += 32) {
    v16h a = load_a_frag(sT + mt * 16 * LDA, lane, kk);
#pragma unroll
    for (int t = 0; t < 4; ++t) {
      v16h b = load_b_frag(sW, lane, nbase + t * 16, kk);
      acc[t] = __builtin_amdgcn_wmma_f32_16x16x32_f16(
          false, a, false, b, (short)0, acc[t], false, false);
    }
  }

  // Fused message + scatter epilogue directly from WMMA accumulators.
  int m_in = (lane >> 4) * 8;
  int nc = lane & 15;
#pragma unroll
  for (int t = 0; t < 4; ++t) {
    int ncol = nbase + t * 16 + nc;
    float bv = be2[ncol];
#pragma unroll
    for (int r = 0; r < 8; ++r) {
      int m = mt * 16 + m_in + r;
      long e = e0 + m;
      if (e < E_EDGES) {
        float filt = acc[t][r] + bv;
        float msg = hin[(long)sSrc[m] * C_CH + ncol] * filt * sEnv[m];
        atomicAdd(&agg[(long)sDst[m] * C_CH + ncol], msg);
      }
    }
  }
}

// ---------------------------------------------------------------------------
// BatchNorm helpers
// ---------------------------------------------------------------------------
__global__ __launch_bounds__(128) void bn_stats_kernel(
    const float* __restrict__ h, float* __restrict__ sums) {
  int c = threadIdx.x;          // one channel per thread
  int n0 = blockIdx.x * 512;
  int nend = n0 + 512; if (nend > N_NODES) nend = N_NODES;
  float s = 0.0f, s2 = 0.0f;
  for (int n = n0; n < nend; ++n) {
    float v = h[(long)n * C_CH + c];
    s += v; s2 += v * v;
  }
  atomicAdd(&sums[c], s);
  atomicAdd(&sums[128 + c], s2);
}

__global__ __launch_bounds__(256) void bn_apply_kernel(
    float* __restrict__ h, const float* __restrict__ sums,
    const float* __restrict__ g, const float* __restrict__ b) {
  long idx = (long)blockIdx.x * 256 + threadIdx.x;
  if (idx >= (long)N_NODES * C_CH) return;
  int c = (int)(idx & 127);
  const float inv_n = 1.0f / (float)N_NODES;
  float mu = sums[c] * inv_n;
  float var = sums[128 + c] * inv_n - mu * mu;
  h[idx] = (h[idx] - mu) * rsqrtf(var + BN_EPS) * g[c] + b[c];
}

// ---------------------------------------------------------------------------
// Graph pooling + readout MLP
// ---------------------------------------------------------------------------
__global__ __launch_bounds__(256) void pool_kernel(
    const float* __restrict__ h, const int* __restrict__ batch,
    float* __restrict__ hg) {
  long idx = (long)blockIdx.x * 256 + threadIdx.x;
  if (idx >= (long)N_NODES * C_CH) return;
  int n = (int)(idx >> 7);
  int c = (int)(idx & 127);
  atomicAdd(&hg[(long)batch[n] * C_CH + c], h[idx]);
}

__global__ __launch_bounds__(32) void readout_kernel(
    const float* __restrict__ hg, const float* __restrict__ W1,
    const float* __restrict__ b1, const float* __restrict__ W2,
    const float* __restrict__ b2, float* __restrict__ out) {
  int g = blockIdx.x;
  int lane = threadIdx.x;  // 32 lanes = 32 hidden units
  const float* row = hg + (long)g * C_CH;
  float acc = b1[lane];
#pragma unroll 4
  for (int k = 0; k < C_CH; ++k) acc += row[k] * W1[k * 32 + lane];
  float t = ssp(acc) * W2[lane];
#pragma unroll
  for (int off = 16; off > 0; off >>= 1) t += __shfl_down(t, off, 32);
  if (lane == 0) out[g] = ssp(t + b2[0]);
}

// ---------------------------------------------------------------------------
extern "C" void kernel_launch(void* const* d_in, const int* in_sizes, int n_in,
                              void* d_out, int out_size, void* d_ws, size_t ws_size,
                              hipStream_t stream) {
  const float* x          = (const float*)d_in[0];
  const int*   edge_index = (const int*)d_in[1];
  const float* edge_w     = (const float*)d_in[2];
  const float* edge_attr  = (const float*)d_in[3];
  const int*   batch      = (const int*)d_in[4];
  const float* W_l1 = (const float*)d_in[5];
  const float* b_l1 = (const float*)d_in[6];
  const float* W_l2 = (const float*)d_in[7];
  const float* b_l2 = (const float*)d_in[8];
  const float* bn_g = (const float*)d_in[9];
  const float* bn_b = (const float*)d_in[10];
  const float* Wi_in  = (const float*)d_in[11];
  const float* bi_in  = (const float*)d_in[12];
  const float* We1    = (const float*)d_in[13];
  const float* be1    = (const float*)d_in[14];
  const float* We2    = (const float*)d_in[15];
  const float* be2    = (const float*)d_in[16];
  const float* Wi_out = (const float*)d_in[17];
  const float* bi_out = (const float*)d_in[18];
  const float* W_r1 = (const float*)d_in[19];
  const float* b_r1 = (const float*)d_in[20];
  const float* W_r2 = (const float*)d_in[21];
  const float* b_r2 = (const float*)d_in[22];
  float* out = (float*)d_out;

  const size_t NC = (size_t)N_NODES * C_CH;
  float* h   = (float*)d_ws;           // [N,C]
  float* hin = h + NC;                 // [N,C] (also embed temp)
  float* agg = hin + NC;               // [N,C]
  float* bns = agg + NC;               // [256] sum / sumsq
  float* hg  = bns + 256;              // [G,C]

  const int gemm_blocks = (N_NODES + 63) / 64;
  const int nc_blocks   = (int)((NC + 255) / 256);

  // Node embedding: Linear(16->128) -> Linear(128->128)
  gemm_bias_ssp_kernel<<<gemm_blocks, 256, 0, stream>>>(
      x, W_l1, b_l1, hin, nullptr, N_NODES, FIN_F, 0);
  gemm_bias_ssp_kernel<<<gemm_blocks, 256, 0, stream>>>(
      hin, W_l2, b_l2, h, nullptr, N_NODES, C_CH, 0);

  // BatchNorm (training-mode batch stats)
  hipMemsetAsync(bns, 0, 256 * sizeof(float), stream);
  bn_stats_kernel<<<(N_NODES + 511) / 512, 128, 0, stream>>>(h, bns);
  bn_apply_kernel<<<nc_blocks, 256, 0, stream>>>(h, bns, bn_g, bn_b);

  // Interaction layers
  for (int l = 0; l < L_LAYERS; ++l) {
    gemm_bias_ssp_kernel<<<gemm_blocks, 256, 0, stream>>>(
        h, Wi_in + (size_t)l * C_CH * C_CH, bi_in + (size_t)l * C_CH,
        hin, nullptr, N_NODES, C_CH, 0);
    hipMemsetAsync(agg, 0, NC * sizeof(float), stream);
    edge_cfconv_kernel<<<(E_EDGES + 63) / 64, 256, 0, stream>>>(
        hin, edge_attr, edge_w, edge_index,
        We1 + (size_t)l * FE_F * C_CH, be1 + (size_t)l * C_CH,
        We2 + (size_t)l * C_CH * C_CH, be2 + (size_t)l * C_CH, agg);
    gemm_bias_ssp_kernel<<<gemm_blocks, 256, 0, stream>>>(
        agg, Wi_out + (size_t)l * C_CH * C_CH, bi_out + (size_t)l * C_CH,
        h, h, N_NODES, C_CH, 1);
  }

  // Readout: segment-sum pool + 2-layer SSP MLP
  hipMemsetAsync(hg, 0, (size_t)G_GRAPHS * C_CH * sizeof(float), stream);
  pool_kernel<<<nc_blocks, 256, 0, stream>>>(h, batch, hg);
  readout_kernel<<<G_GRAPHS, 32, 0, stream>>>(hg, W_r1, b_r1, W_r2, b_r2, out);
}
